// AttentionBlock_82738249990824
// MI455X (gfx1250) — matrix operand order repaired and checked by reading, not compiled, not run on hardware
//
#include <hip/hip_runtime.h>

#define NPIX 16384   // H*W
#define CCH  64      // channels
#define LOG2E 1.4426950408889634f

typedef __attribute__((ext_vector_type(2)))  float    v2f;
typedef __attribute__((ext_vector_type(8)))  float    v8f;
typedef __attribute__((ext_vector_type(2)))  __fp16   v2fp16;
typedef __attribute__((ext_vector_type(8)))  _Float16 v8h;
typedef __attribute__((ext_vector_type(16))) _Float16 v16h;

static __device__ __forceinline__ unsigned pack2h(float a, float b) {
  // single v_cvt_pk_rtz_f16_f32
  union { v2fp16 h; unsigned u; } t;
  t.h = __builtin_amdgcn_cvt_pkrtz(a, b);
  return t.u;
}

static __device__ __forceinline__ v8f wmma_f32_k4(v2f a, v2f b, v8f c) {
  return __builtin_amdgcn_wmma_f32_16x16x4_f32(
      false, a, false, b, (short)0, c, false, false);
}

// ---------------------------------------------------------------------------
// Kernel 1: projections.  One thread per pixel n.
//   q[n][:] = LOG2E * (Wq x[:,n] + bq)   -> logits arrive in log2 domain
//   k[n][:] =          Wk x[:,n] + bk
//   vT[c][n] = f16(Wv x[:,n] + bv)       (channel-major for WMMA A-operand)
// ---------------------------------------------------------------------------
__global__ __launch_bounds__(256) void proj_kernel(
    const float* __restrict__ x,
    const float* __restrict__ Wq, const float* __restrict__ bq,
    const float* __restrict__ Wk, const float* __restrict__ bk,
    const float* __restrict__ Wv, const float* __restrict__ bv,
    float* __restrict__ qo, float* __restrict__ ko,
    _Float16* __restrict__ vT)
{
  const int n = blockIdx.x * blockDim.x + threadIdx.x;
  float qa[8], ka[8], va[64];
#pragma unroll
  for (int d = 0; d < 8; ++d) { qa[d] = bq[d]; ka[d] = bk[d]; }
#pragma unroll
  for (int j = 0; j < 64; ++j) va[j] = bv[j];

  for (int c = 0; c < CCH; ++c) {
    const float xc = x[c * NPIX + n];
#pragma unroll
    for (int d = 0; d < 8; ++d) {
      qa[d] = fmaf(Wq[d * CCH + c], xc, qa[d]);
      ka[d] = fmaf(Wk[d * CCH + c], xc, ka[d]);
    }
#pragma unroll
    for (int j = 0; j < 64; ++j)
      va[j] = fmaf(Wv[j * CCH + c], xc, va[j]);
  }
#pragma unroll
  for (int d = 0; d < 8; ++d) {
    qo[n * 8 + d] = qa[d] * LOG2E;   // fold log2(e) into Q once
    ko[n * 8 + d] = ka[d];
  }
#pragma unroll
  for (int j = 0; j < 64; ++j) vT[j * NPIX + n] = (_Float16)va[j];
}

// ---------------------------------------------------------------------------
// Kernel 2: streaming attention, transposed formulation, 2 query tiles/wave.
//   S^T = K * Q^T        (v_wmma_f32_16x16x4_f32, exact f32 logits)
//   online softmax       (raw v_exp_f32; stats are per-lane scalars)
//   O^T += V^T * P^T     (v_wmma_f32_16x16x32_f16, K/V operands shared
//                         across both query tiles)
// ---------------------------------------------------------------------------
__global__ __launch_bounds__(256) void attn_kernel(
    const float* __restrict__ qbuf, const float* __restrict__ kbuf,
    const _Float16* __restrict__ vT,
    const float* __restrict__ x, const float* __restrict__ gamma,
    float* __restrict__ out)
{
  const int lane  = threadIdx.x & 31;
  const int wid   = threadIdx.x >> 5;
  const int pair  = blockIdx.x * 8 + wid;   // 0..511
  const int qbase = pair * 32;              // 32 queries per wave
  const int h     = lane >> 4;
  const int col   = lane & 15;
  const int foff  = h ? 2 : 0;

  // Q as B-operand (4x16 f32) for both query tiles
  v2f bq0[2], bq1[2];
#pragma unroll
  for (int t = 0; t < 2; ++t) {
    const float* qp = qbuf + (size_t)(qbase + 16 * t + col) * 8;
    bq0[t] = *(const v2f*)(qp + foff);
    bq1[t] = *(const v2f*)(qp + 4 + foff);
  }

  float m[2] = {-3.0e38f, -3.0e38f};
  float l[2] = {0.0f, 0.0f};
  v8f o[2][4];
#pragma unroll
  for (int t = 0; t < 2; ++t)
#pragma unroll
    for (int j = 0; j < 4; ++j) o[t][j] = (v8f){};

  for (int kb = 0; kb < NPIX; kb += 32) {
    // --- K tile as A-operand (shared by both query tiles) ---
    const float* kp0 = kbuf + (size_t)(kb + col) * 8;
    const float* kp1 = kp0 + 16 * 8;
    const v2f ak00 = *(const v2f*)(kp0 + foff);
    const v2f ak01 = *(const v2f*)(kp0 + 4 + foff);
    const v2f ak10 = *(const v2f*)(kp1 + foff);
    const v2f ak11 = *(const v2f*)(kp1 + 4 + foff);
    const v8f z = (v8f){};

    union { unsigned u[8]; v16h v; } pb[2];

#pragma unroll
    for (int t = 0; t < 2; ++t) {
      // logits (already log2-scaled via Q)
      v8f slo = wmma_f32_k4(ak01, bq1[t], wmma_f32_k4(ak00, bq0[t], z));
      v8f shi = wmma_f32_k4(ak11, bq1[t], wmma_f32_k4(ak10, bq0[t], z));

      // online softmax; query index == lane column -> per-lane scalar stats
      float tmax = -3.0e38f;
#pragma unroll
      for (int r = 0; r < 8; ++r) {
        tmax = fmaxf(tmax, slo[r]);
        tmax = fmaxf(tmax, shi[r]);
      }
      tmax = fmaxf(tmax, __shfl_xor(tmax, 16, 32));
      const float mn    = fmaxf(m[t], tmax);
      const float alpha = __builtin_amdgcn_exp2f(m[t] - mn);  // raw v_exp_f32

      float plo[8], phi[8];
      float sum = 0.0f;
#pragma unroll
      for (int r = 0; r < 8; ++r) {
        plo[r] = __builtin_amdgcn_exp2f(slo[r] - mn);
        phi[r] = __builtin_amdgcn_exp2f(shi[r] - mn);
        sum += plo[r] + phi[r];
      }
      sum += __shfl_xor(sum, 16, 32);
      l[t] = l[t] * alpha + sum;
      m[t] = mn;
#pragma unroll
      for (int j = 0; j < 4; ++j)
#pragma unroll
        for (int r = 0; r < 8; ++r) o[t][j][r] *= alpha;

      // P^T as f16 B-operand: pack pairs, one half-wave exchange
      unsigned lo_pk[4], hi_pk[4];
#pragma unroll
      for (int i = 0; i < 4; ++i) {
        lo_pk[i] = pack2h(plo[2 * i], plo[2 * i + 1]);
        hi_pk[i] = pack2h(phi[2 * i], phi[2 * i + 1]);
      }
#pragma unroll
      for (int i = 0; i < 4; ++i) {
        const unsigned lo_x = (unsigned)__shfl_xor((int)lo_pk[i], 16, 32);
        const unsigned hi_x = (unsigned)__shfl_xor((int)hi_pk[i], 16, 32);
        pb[t].u[i]     = h ? hi_x     : lo_pk[i];
        pb[t].u[4 + i] = h ? hi_pk[i] : lo_x;
      }
    }

    // --- O^T += V^T * P^T : V operands shared across both query tiles ---
    const int koff = kb + (h ? 16 : 0);
#pragma unroll
    for (int j = 0; j < 4; ++j) {
      const _Float16* vp = vT + (size_t)(16 * j + col) * NPIX + koff;
      union { v8h p[2]; v16h v; } va;
      va.p[0] = *(const v8h*)(vp);
      va.p[1] = *(const v8h*)(vp + 8);
      o[0][j] = __builtin_amdgcn_wmma_f32_16x16x32_f16(
          false, va.v, false, pb[0].v, (short)0, o[0][j], false, false);
      o[1][j] = __builtin_amdgcn_wmma_f32_16x16x32_f16(
          false, va.v, false, pb[1].v, (short)0, o[1][j], false, false);
    }
  }

  // --- epilogue: out[c][n] = gamma * O[n][c]/l + x[c][n] ---
  const float g = gamma[0];
#pragma unroll
  for (int t = 0; t < 2; ++t) {
    const float invl = 1.0f / l[t];
    const int   nq   = qbase + 16 * t + col;
#pragma unroll
    for (int j = 0; j < 4; ++j)
#pragma unroll
      for (int r = 0; r < 8; ++r) {
        const int    c   = 16 * j + r + 8 * h;
        const size_t idx = (size_t)c * NPIX + nq;
        out[idx] = fmaf(g, o[t][j][r] * invl, x[idx]);
      }
  }
}

// ---------------------------------------------------------------------------
extern "C" void kernel_launch(void* const* d_in, const int* in_sizes, int n_in,
                              void* d_out, int out_size, void* d_ws, size_t ws_size,
                              hipStream_t stream) {
  const float* x     = (const float*)d_in[0];
  const float* Wq    = (const float*)d_in[1];
  const float* bq    = (const float*)d_in[2];
  const float* Wk    = (const float*)d_in[3];
  const float* bk    = (const float*)d_in[4];
  const float* Wv    = (const float*)d_in[5];
  const float* bv    = (const float*)d_in[6];
  const float* gamma = (const float*)d_in[7];
  float* out = (float*)d_out;

  float*    qbuf = (float*)d_ws;                 // [N][8]  f32
  float*    kbuf = qbuf + NPIX * 8;              // [N][8]  f32
  _Float16* vT   = (_Float16*)(kbuf + NPIX * 8); // [64][N] f16

  proj_kernel<<<NPIX / 256, 256, 0, stream>>>(x, Wq, bq, Wk, bk, Wv, bv,
                                              qbuf, kbuf, vT);
  attn_kernel<<<64, 256, 0, stream>>>(qbuf, kbuf, vT, x, gamma, out);
}
